// MoELayer_83932250899016
// MI455X (gfx1250) — matrix operand order repaired and checked by reading, not compiled, not run on hardware
//
#include <hip/hip_runtime.h>
#include <hip/hip_bf16.h>

// ---------------------------------------------------------------------------
// MoE (HID=1024, FFN=4096, E=16, top-4) for MI455X / gfx1250.
// Sparse grouped-GEMM; one-shot fp32->bf16 conversion (weights transposed to
// [N][K]), GEMM main loops stream tiles with GLOBAL_LOAD_ASYNC_TO_LDS_B128
// (ASYNCcnt) straight into LDS, compute with v_wmma_f32_16x16x32_bf16.
// ---------------------------------------------------------------------------

typedef __attribute__((ext_vector_type(16))) __bf16 v16bf;
typedef __attribute__((ext_vector_type(8)))  float  v8f;

constexpr int HID   = 1024;
constexpr int FFN   = 4096;
constexpr int NE    = 16;
constexpr int TOPK  = 4;
constexpr int T_TOK = 4 * 2048;              // B*S = 8192 tokens
constexpr int R_RT  = T_TOK * TOPK;          // 32768 routed rows
constexpr int KC    = 64;                    // K-chunk (two 16x16x32 steps)
constexpr int KP    = 72;                    // padded LDS row (bf16), 144B = 16B-aligned

// ---- workspace layout (bytes) ----
constexpr size_t OFF_COUNTS = 0;                                          // 16 int
constexpr size_t OFF_CURSOR = 256;                                        // 16 int
constexpr size_t OFF_OFFSET = 512;                                        // 16 int
constexpr size_t OFF_RE     = 1024;                                       // R int
constexpr size_t OFF_RW     = OFF_RE    + (size_t)R_RT * 4;               // R float
constexpr size_t OFF_TOK    = OFF_RW    + (size_t)R_RT * 4;               // R int
constexpr size_t OFF_WROW   = OFF_TOK   + (size_t)R_RT * 4;               // R float
constexpr size_t OFF_ENTRY  = OFF_WROW  + (size_t)R_RT * 4;               // R int
constexpr size_t OFF_XBF    = (OFF_ENTRY + (size_t)R_RT * 4 + 255) & ~(size_t)255;
constexpr size_t OFF_W1T    = OFF_XBF  + (size_t)T_TOK * HID * 2;         // bf16 [E][FFN][HID]
constexpr size_t OFF_W2T    = OFF_W1T  + (size_t)NE * HID * FFN * 2;      // bf16 [E][HID][FFN]
constexpr size_t OFF_HDN    = OFF_W2T  + (size_t)NE * FFN * HID * 2;      // bf16 [R][FFN]
constexpr size_t OFF_YBUF   = OFF_HDN  + (size_t)R_RT * FFN * 2;          // f32  [R][HID]
constexpr size_t WS_NEED    = OFF_YBUF + (size_t)R_RT * HID * 4;

__device__ __forceinline__ unsigned short f32_to_bf16(float f) {
    unsigned int u = __float_as_uint(f);
    unsigned int r = u + 0x7FFFu + ((u >> 16) & 1u);   // round-to-nearest-even
    return (unsigned short)(r >> 16);
}

union Frag { uint4 q[2]; v16bf v; };

__device__ __forceinline__ v8f wmma_bf16(v16bf a, v16bf b, v8f c) {
    return __builtin_amdgcn_wmma_f32_16x16x32_bf16(
        false, a, false, b, (short)0, c, false, false);
}

// Async 16B/lane global -> LDS copy (ASYNCcnt-tracked, no VGPR round trip).
// lds = per-lane LDS byte address (low 32 bits of the flat address), g = per-lane
// global source.
__device__ __forceinline__ void async_g2l_b128(unsigned lds, const void* g) {
    asm volatile("global_load_async_to_lds_b128 %0, %1, off"
                 :: "v"(lds), "v"(g) : "memory");
}
__device__ __forceinline__ void wait_async0() {
    asm volatile("s_wait_asynccnt 0" ::: "memory");
}

// A fragment (16x32 bf16): lane l -> row mbase+(l&15); K runs [h*8,+8) and
// [16+h*8,+8), h=l>>4  ==> two 16B LDS loads.
__device__ __forceinline__ v16bf frag_a(const unsigned short (*As)[KP], int mbase,
                                        int ks, int lane) {
    const unsigned short* p = &As[mbase + (lane & 15)][ks * 32 + (lane >> 4) * 8];
    Frag f;
    f.q[0] = *(const uint4*)p;
    f.q[1] = *(const uint4*)(p + 16);
    return f.v;
}
// B fragment (32x16 bf16), Bs stored [n][k]: lane l -> col nbase+(l&15);
// K runs [h*8,+8) and [16+h*8,+8), h=l>>4  ==> two 16B LDS loads.
__device__ __forceinline__ v16bf frag_b(const unsigned short (*Bs)[KP], int nbase,
                                        int ks, int lane) {
    const unsigned short* p = &Bs[nbase + (lane & 15)][ks * 32 + (lane >> 4) * 8];
    Frag f;
    f.q[0] = *(const uint4*)p;
    f.q[1] = *(const uint4*)(p + 16);
    return f.v;
}

__device__ __forceinline__ float gelu_exact(float x) {
    return 0.5f * x * (1.0f + erff(x * 0.70710678118654752f));
}

// ---------------------------------------------------------------------------
__global__ void moe_init(int* counts) {
    if (threadIdx.x < NE) counts[threadIdx.x] = 0;
}

// One wave per token: logits = x . Wg + bg ; top-4 ; softmax.
__global__ __launch_bounds__(32) void moe_gate(
    const float* __restrict__ x, const float* __restrict__ Wg,
    const float* __restrict__ bg, int* __restrict__ routeE,
    float* __restrict__ routeW, int* __restrict__ counts) {
    const int t = blockIdx.x;
    const int lane = threadIdx.x;
    const float* xt = x + (size_t)t * HID;

    float lg[NE];
#pragma unroll
    for (int e = 0; e < NE; ++e) lg[e] = 0.f;

    for (int h = lane; h < HID; h += 32) {
        const float xv = xt[h];
        const float4* wr = (const float4*)(Wg + (size_t)h * NE);
        float w[NE];
        *(float4*)&w[0]  = wr[0];
        *(float4*)&w[4]  = wr[1];
        *(float4*)&w[8]  = wr[2];
        *(float4*)&w[12] = wr[3];
#pragma unroll
        for (int e = 0; e < NE; ++e) lg[e] += xv * w[e];
    }
#pragma unroll
    for (int e = 0; e < NE; ++e) {
        float v = lg[e];
#pragma unroll
        for (int m = 16; m >= 1; m >>= 1) v += __shfl_xor(v, m, 32);
        lg[e] = v;
    }
    if (lane == 0) {
#pragma unroll
        for (int e = 0; e < NE; ++e) lg[e] += bg[e];
        int idx[TOPK]; float val[TOPK];
        unsigned used = 0;
#pragma unroll
        for (int k = 0; k < TOPK; ++k) {
            float best = -3.4e38f; int bi = 0;
#pragma unroll
            for (int e = 0; e < NE; ++e)
                if (!((used >> e) & 1u) && lg[e] > best) { best = lg[e]; bi = e; }
            used |= (1u << bi); idx[k] = bi; val[k] = best;
        }
        const float mx = val[0];
        float w[TOPK], s = 0.f;
#pragma unroll
        for (int k = 0; k < TOPK; ++k) { w[k] = __expf(val[k] - mx); s += w[k]; }
        const float inv = 1.0f / s;
#pragma unroll
        for (int k = 0; k < TOPK; ++k) {
            routeE[t * TOPK + k] = idx[k];
            routeW[t * TOPK + k] = w[k] * inv;
            atomicAdd(&counts[idx[k]], 1);
        }
    }
}

__global__ void moe_scan(const int* counts, int* offsets, int* cursor) {
    if (threadIdx.x == 0) {
        int off = 0;
        for (int e = 0; e < NE; ++e) {
            offsets[e] = off; cursor[e] = off; off += counts[e];
        }
    }
}

__global__ void moe_scatter(const int* __restrict__ routeE,
                            const float* __restrict__ routeW,
                            int* cursor, int* __restrict__ tok_of,
                            float* __restrict__ w_of, int* __restrict__ entry_of) {
    const int i = blockIdx.x * blockDim.x + threadIdx.x;
    if (i >= R_RT) return;
    const int e = routeE[i];
    const int pos = atomicAdd(&cursor[e], 1);
    tok_of[pos] = i >> 2;
    w_of[pos] = routeW[i];
    entry_of[i] = pos;
}

// x fp32 -> bf16 (elementwise, float4 granules).
__global__ void moe_cvt_x(const float* __restrict__ x,
                          unsigned short* __restrict__ xbf) {
    const int i = blockIdx.x * blockDim.x + threadIdx.x;  // < T*HID/4
    const float4 f = ((const float4*)x)[i];
    ushort4 u;
    u.x = f32_to_bf16(f.x); u.y = f32_to_bf16(f.y);
    u.z = f32_to_bf16(f.z); u.w = f32_to_bf16(f.w);
    ((ushort4*)xbf)[i] = u;
}

// W [E][K][N] fp32  ->  Wt [E][N][K] bf16  (64x64 tiles via LDS).
__global__ __launch_bounds__(256) void moe_cvt_wt(const float* __restrict__ W,
                                                  unsigned short* __restrict__ Wt,
                                                  int K, int N) {
    __shared__ unsigned short t[64][72];
    const int k0 = blockIdx.x * 64, n0 = blockIdx.y * 64, e = blockIdx.z;
    const float* We = W + (size_t)e * K * N;
    unsigned short* Wte = Wt + (size_t)e * K * N;
    const int tid = threadIdx.x;
#pragma unroll
    for (int j = 0; j < 4; ++j) {
        const int v = tid + j * 256;          // 64 k-rows x 16 n-quads
        const int k = v >> 4, nq = v & 15;
        const float4 f = *(const float4*)(We + (size_t)(k0 + k) * N + n0 + nq * 4);
        ushort4 u;
        u.x = f32_to_bf16(f.x); u.y = f32_to_bf16(f.y);
        u.z = f32_to_bf16(f.z); u.w = f32_to_bf16(f.w);
        *(ushort4*)&t[k][nq * 4] = u;
    }
    __syncthreads();
#pragma unroll
    for (int j = 0; j < 2; ++j) {
        const int v = tid + j * 256;          // 64 n-rows x 8 k-octets
        const int n = v >> 3, ko = (v & 7) * 8;
        ushort4 a, b;
        a.x = t[ko + 0][n]; a.y = t[ko + 1][n]; a.z = t[ko + 2][n]; a.w = t[ko + 3][n];
        b.x = t[ko + 4][n]; b.y = t[ko + 5][n]; b.z = t[ko + 6][n]; b.w = t[ko + 7][n];
        unsigned short* o = Wte + (size_t)(n0 + n) * K + k0 + ko;
        *(ushort4*)o = a;
        *(ushort4*)(o + 4) = b;
    }
}

// ---------------------------------------------------------------------------
// FFN1: hdn[row, :] = gelu(xbf[tok(row)] @ W1t^T + b1[e])   (bf16 out)
// 128x128 macro tile, 8 waves x (64x32 = 4x2 WMMA accum tiles), K = 1024.
__global__ __launch_bounds__(256) void moe_ffn1(
    const unsigned short* __restrict__ xbf, const unsigned short* __restrict__ W1t,
    const float* __restrict__ b1, const int* __restrict__ counts,
    const int* __restrict__ offsets, const int* __restrict__ tok_of,
    unsigned short* __restrict__ hdn) {
    __shared__ unsigned short As[128][KP];
    __shared__ unsigned short Bs[128][KP];

    const int e = blockIdx.z;
    const int cnt = counts[e];
    const int mt0 = blockIdx.x * 128;
    if (mt0 >= cnt) return;
    const int rstart = offsets[e];
    const int n0 = blockIdx.y * 128;
    const unsigned short* W1te = W1t + (size_t)e * FFN * HID;  // [FFN][HID]

    const int tid = threadIdx.x, lane = tid & 31, wave = tid >> 5;
    const int waveM = wave & 1, waveN = wave >> 1;
    const int half = lane >> 4, ln = lane & 15;

    const unsigned as_base = (unsigned)(uintptr_t)&As[0][0];
    const unsigned bs_base = (unsigned)(uintptr_t)&Bs[0][0];

    // Per-thread transfer descriptors (4 A + 4 B, 16B each), fixed across K.
    const unsigned short* ga[4]; unsigned la[4];
    const unsigned short* gb[4]; unsigned lb[4];
#pragma unroll
    for (int j = 0; j < 4; ++j) {
        const int v = tid + j * 256;          // 128 rows x 8 16B-chunks
        const int r = v >> 3, c = v & 7;
        const int rl = mt0 + r;
        int g = rstart + rl; if (g > R_RT - 1) g = R_RT - 1;
        const int tok = (rl < cnt) ? tok_of[g] : 0;
        ga[j] = xbf + (size_t)tok * HID + c * 8;
        la[j] = as_base + (unsigned)(r * KP + c * 8) * 2;
        gb[j] = W1te + (size_t)(n0 + r) * HID + c * 8;
        lb[j] = bs_base + (unsigned)(r * KP + c * 8) * 2;
    }

    v8f acc[4][2];
#pragma unroll
    for (int mt = 0; mt < 4; ++mt)
#pragma unroll
        for (int nt = 0; nt < 2; ++nt)
#pragma unroll
            for (int i = 0; i < 8; ++i) acc[mt][nt][i] = 0.f;

    for (int k0 = 0; k0 < HID; k0 += KC) {
        __syncthreads();
#pragma unroll
        for (int j = 0; j < 4; ++j) async_g2l_b128(la[j], ga[j] + k0);
#pragma unroll
        for (int j = 0; j < 4; ++j) async_g2l_b128(lb[j], gb[j] + k0);
        if (k0 + KC < HID)
            __builtin_prefetch(gb[0] + k0 + KC, 0, 1);
        wait_async0();
        __syncthreads();

#pragma unroll
        for (int ks = 0; ks < 2; ++ks) {
            const v16bf b0 = frag_b(Bs, waveN * 32, ks, lane);
            const v16bf b1f = frag_b(Bs, waveN * 32 + 16, ks, lane);
#pragma unroll
            for (int mt = 0; mt < 4; ++mt) {
                const v16bf a = frag_a(As, waveM * 64 + mt * 16, ks, lane);
                acc[mt][0] = wmma_bf16(a, b0, acc[mt][0]);
                acc[mt][1] = wmma_bf16(a, b1f, acc[mt][1]);
            }
        }
    }

    const float* b1e = b1 + (size_t)e * FFN;
#pragma unroll
    for (int mt = 0; mt < 4; ++mt) {
        const int rl0 = mt0 + waveM * 64 + mt * 16 + half * 8;
#pragma unroll
        for (int nt = 0; nt < 2; ++nt) {
            const int col = n0 + waveN * 32 + nt * 16 + ln;
            const float bias = b1e[col];
#pragma unroll
            for (int v = 0; v < 8; ++v) {
                const int row = rl0 + v;
                if (row < cnt) {
                    const float val = acc[mt][nt][v] + bias;
                    hdn[(size_t)(rstart + row) * FFN + col] = f32_to_bf16(gelu_exact(val));
                }
            }
        }
    }
}

// ---------------------------------------------------------------------------
// FFN2: ybuf[row, :] = w(row) * (hdn[row] @ W2t^T + b2[e])   (f32 out), K = 4096.
__global__ __launch_bounds__(256) void moe_ffn2(
    const unsigned short* __restrict__ hdn, const unsigned short* __restrict__ W2t,
    const float* __restrict__ b2, const int* __restrict__ counts,
    const int* __restrict__ offsets, const float* __restrict__ w_of,
    float* __restrict__ ybuf) {
    __shared__ unsigned short As[128][KP];
    __shared__ unsigned short Bs[128][KP];

    const int e = blockIdx.z;
    const int cnt = counts[e];
    const int mt0 = blockIdx.x * 128;
    if (mt0 >= cnt) return;
    const int rstart = offsets[e];
    const int n0 = blockIdx.y * 128;
    const unsigned short* W2te = W2t + (size_t)e * HID * FFN;  // [HID][FFN]

    const int tid = threadIdx.x, lane = tid & 31, wave = tid >> 5;
    const int waveM = wave & 1, waveN = wave >> 1;
    const int half = lane >> 4, ln = lane & 15;

    const unsigned as_base = (unsigned)(uintptr_t)&As[0][0];
    const unsigned bs_base = (unsigned)(uintptr_t)&Bs[0][0];

    const unsigned short* ga[4]; unsigned la[4];
    const unsigned short* gb[4]; unsigned lb[4];
#pragma unroll
    for (int j = 0; j < 4; ++j) {
        const int v = tid + j * 256;
        const int r = v >> 3, c = v & 7;
        int g = rstart + mt0 + r; if (g > R_RT - 1) g = R_RT - 1;
        ga[j] = hdn + (size_t)g * FFN + c * 8;
        la[j] = as_base + (unsigned)(r * KP + c * 8) * 2;
        gb[j] = W2te + (size_t)(n0 + r) * FFN + c * 8;
        lb[j] = bs_base + (unsigned)(r * KP + c * 8) * 2;
    }

    v8f acc[4][2];
#pragma unroll
    for (int mt = 0; mt < 4; ++mt)
#pragma unroll
        for (int nt = 0; nt < 2; ++nt)
#pragma unroll
            for (int i = 0; i < 8; ++i) acc[mt][nt][i] = 0.f;

    for (int k0 = 0; k0 < FFN; k0 += KC) {
        __syncthreads();
#pragma unroll
        for (int j = 0; j < 4; ++j) async_g2l_b128(la[j], ga[j] + k0);
#pragma unroll
        for (int j = 0; j < 4; ++j) async_g2l_b128(lb[j], gb[j] + k0);
        if (k0 + KC < FFN)
            __builtin_prefetch(gb[0] + k0 + KC, 0, 1);
        wait_async0();
        __syncthreads();

#pragma unroll
        for (int ks = 0; ks < 2; ++ks) {
            const v16bf b0 = frag_b(Bs, waveN * 32, ks, lane);
            const v16bf b1f = frag_b(Bs, waveN * 32 + 16, ks, lane);
#pragma unroll
            for (int mt = 0; mt < 4; ++mt) {
                const v16bf a = frag_a(As, waveM * 64 + mt * 16, ks, lane);
                acc[mt][0] = wmma_bf16(a, b0, acc[mt][0]);
                acc[mt][1] = wmma_bf16(a, b1f, acc[mt][1]);
            }
        }
    }

    const float* b2e = b2 + (size_t)e * HID;
#pragma unroll
    for (int mt = 0; mt < 4; ++mt) {
        const int rl0 = mt0 + waveM * 64 + mt * 16 + half * 8;
#pragma unroll
        for (int nt = 0; nt < 2; ++nt) {
            const int col = n0 + waveN * 32 + nt * 16 + ln;
            const float bias = b2e[col];
#pragma unroll
            for (int v = 0; v < 8; ++v) {
                const int row = rl0 + v;
                if (row < cnt) {
                    const int g = rstart + row;
                    ybuf[(size_t)g * HID + col] = w_of[g] * (acc[mt][nt][v] + bias);
                }
            }
        }
    }
}

// Deterministic combine: out[t,c] = sum_k ybuf[entry_of[t,k], c]
__global__ void moe_combine(const float* __restrict__ ybuf,
                            const int* __restrict__ entry_of,
                            float* __restrict__ out) {
    const int i = blockIdx.x * blockDim.x + threadIdx.x;  // < T_TOK*HID
    const int t = i >> 10, c = i & (HID - 1);
    const int* ent = entry_of + t * TOPK;
    float s = ybuf[(size_t)ent[0] * HID + c];
    s += ybuf[(size_t)ent[1] * HID + c];
    s += ybuf[(size_t)ent[2] * HID + c];
    s += ybuf[(size_t)ent[3] * HID + c];
    out[i] = s;
}

// ---------------------------------------------------------------------------
extern "C" void kernel_launch(void* const* d_in, const int* in_sizes, int n_in,
                              void* d_out, int out_size, void* d_ws, size_t ws_size,
                              hipStream_t stream) {
    const float* x  = (const float*)d_in[0];
    const float* Wg = (const float*)d_in[1];
    const float* bg = (const float*)d_in[2];
    const float* W1 = (const float*)d_in[3];
    const float* b1 = (const float*)d_in[4];
    const float* W2 = (const float*)d_in[5];
    const float* b2 = (const float*)d_in[6];
    float* out = (float*)d_out;
    char* ws = (char*)d_ws;
    if (ws_size < WS_NEED) return;  // ~657 MB scratch required

    int*   counts   = (int*)(ws + OFF_COUNTS);
    int*   cursor   = (int*)(ws + OFF_CURSOR);
    int*   offsets  = (int*)(ws + OFF_OFFSET);
    int*   routeE   = (int*)(ws + OFF_RE);
    float* routeW   = (float*)(ws + OFF_RW);
    int*   tok_of   = (int*)(ws + OFF_TOK);
    float* w_of     = (float*)(ws + OFF_WROW);
    int*   entry_of = (int*)(ws + OFF_ENTRY);
    unsigned short* xbf = (unsigned short*)(ws + OFF_XBF);
    unsigned short* W1t = (unsigned short*)(ws + OFF_W1T);
    unsigned short* W2t = (unsigned short*)(ws + OFF_W2T);
    unsigned short* hdn = (unsigned short*)(ws + OFF_HDN);
    float* ybuf     = (float*)(ws + OFF_YBUF);

    moe_init<<<1, 64, 0, stream>>>(counts);
    moe_gate<<<T_TOK, 32, 0, stream>>>(x, Wg, bg, routeE, routeW, counts);
    moe_scan<<<1, 32, 0, stream>>>(counts, offsets, cursor);
    moe_scatter<<<R_RT / 256, 256, 0, stream>>>(routeE, routeW, cursor,
                                                tok_of, w_of, entry_of);
    moe_cvt_x<<<(T_TOK * HID / 4) / 256, 256, 0, stream>>>(x, xbf);
    moe_cvt_wt<<<dim3(HID / 64, FFN / 64, NE), 256, 0, stream>>>(W1, W1t, HID, FFN);
    moe_cvt_wt<<<dim3(FFN / 64, HID / 64, NE), 256, 0, stream>>>(W2, W2t, FFN, HID);
    moe_ffn1<<<dim3(R_RT / 128, FFN / 128, NE), 256, 0, stream>>>(
        xbf, W1t, b1, counts, offsets, tok_of, hdn);
    moe_ffn2<<<dim3(R_RT / 128, HID / 128, NE), 256, 0, stream>>>(
        hdn, W2t, b2, counts, offsets, w_of, ybuf);
    moe_combine<<<(T_TOK * HID) / 256, 256, 0, stream>>>(ybuf, entry_of, out);
}
